// MultiHeadAttention_13185549598974
// MI455X (gfx1250) — matrix operand order repaired and checked by reading, not compiled, not run on hardware
//
#include <hip/hip_runtime.h>
#include <hip/hip_bf16.h>

#define Bv 2
#define Sv 2048
#define Dv 1024
#define Hv 16
#define HDv 64

typedef __attribute__((ext_vector_type(16))) __bf16 v16bf;
typedef __attribute__((ext_vector_type(8)))  __bf16 v8bf;
typedef __attribute__((ext_vector_type(8)))  float  v8f;
typedef unsigned int u32x4 __attribute__((ext_vector_type(4)));
typedef int i32x4 __attribute__((ext_vector_type(4)));
typedef int i32x8 __attribute__((ext_vector_type(8)));

__device__ __forceinline__ unsigned short f2bf(float f) {
  unsigned int u = __builtin_bit_cast(unsigned int, f);
  u += 0x7FFFu + ((u >> 16) & 1u);          // round-to-nearest-even
  return (unsigned short)(u >> 16);
}

__device__ __forceinline__ v8f wmma_bf16(v16bf a, v16bf b, v8f c) {
  return __builtin_amdgcn_wmma_f32_16x16x32_bf16(false, a, false, b, (short)0, c,
                                                 false, false);
}

__device__ __forceinline__ v16bf combine16(v8bf lo, v8bf hi) {
  return __builtin_shufflevector(lo, hi, 0, 1, 2, 3, 4, 5, 6, 7, 8, 9, 10, 11,
                                 12, 13, 14, 15);
}

// A fragment (16xK, 16-bit): lane<16 -> M=lane, kbase=0 ; lane>=16 -> M=lane-16, kbase=16
// elements 0..15 = K kbase..kbase+15
__device__ __forceinline__ v16bf load_a_global(const unsigned short* p, int ld,
                                               int row, int k0, int lane) {
  const unsigned short* q =
      p + (size_t)(row + (lane & 15)) * ld + k0 + ((lane >> 4) << 4);
  return *reinterpret_cast<const v16bf*>(q);
}

// B fragment (32x16, 16-bit): lane<16 -> N=lane, K=k0+0..15 ; lane>=16 -> N=lane-16, K=k0+16..31
// source is N x K row-major (row n holds the K-vector of output column n)
__device__ __forceinline__ v16bf load_b_rowk(const unsigned short* p, int ld,
                                             int ncol, int k0, int lane) {
  const unsigned short* q =
      p + (size_t)(ncol + (lane & 15)) * ld + k0 + ((lane >> 4) << 4);
  return *reinterpret_cast<const v16bf*>(q);
}

// ---------------------------------------------------------------- TDM: 32x64 bf16 tile -> LDS
// D# per CDNA5 ISA ch.8: 2-D tensor, data_size=2B, tile (64,32), row stride 64 elems,
// LDS pad: after every 32 DWORDs (one 128B row) add 8 DWORDs (32B) -> 160B LDS row stride.
#define KSTR 80  // K-tile LDS row stride (elems): 160B rows, 32B aligned (matches TDM pad)
#define VSTR 48  // V^T LDS row stride: 96B rows, 32B aligned
#define PSTR 40  // P LDS row stride: 80B rows, 16B aligned

__device__ __forceinline__ void tdm_load_k_tile(const unsigned short* gsrc,
                                                unsigned int lds_off) {
  const unsigned long long ga = (unsigned long long)(uintptr_t)gsrc;
  u32x4 g0;
  g0[0] = 1u;                                  // count=1 (valid user descriptor)
  g0[1] = lds_off;                             // lds_addr (bytes)
  g0[2] = (unsigned int)ga;                    // global_addr[31:0]
  g0[3] = (unsigned int)((ga >> 32) & 0x01FFFFFFu) | 0x80000000u;  // [56:32] | type=2
  i32x8 g1;
  g1[0] = (int)0x0F110000u;   // data_size=1(2B) | pad_enable | pad_interval=4 | pad_amount=7
  g1[1] = (int)(64u << 16);   // tensor_dim0[15:0] = 64 (in bits 63:48)
  g1[2] = (int)(2048u << 16); // tensor_dim1[15:0] = 2048 (in bits 95:80)
  g1[3] = (int)(64u << 16);   // tile_dim0 = 64 (bits 127:112)
  g1[4] = 32;                 // tile_dim1 = 32 (bits 143:128)
  g1[5] = 64;                 // tensor_dim0_stride = 64 (bits 207:160, low word)
  g1[6] = 0;
  g1[7] = 0;
  i32x4 gz = {0, 0, 0, 0};
#if __clang_major__ >= 23
  i32x8 gz8 = {0, 0, 0, 0, 0, 0, 0, 0};
  __builtin_amdgcn_tensor_load_to_lds(g0, g1, gz, gz, gz8, 0);
#else
  __builtin_amdgcn_tensor_load_to_lds(g0, g1, gz, gz, 0);
#endif
}

// ---------------------------------------------------------------- convert f32 -> bf16 (x4)
__global__ __launch_bounds__(256) void cvt_bf16(const float* __restrict__ src,
                                                unsigned short* __restrict__ dst,
                                                int n4) {
  int i = blockIdx.x * 256 + threadIdx.x;
  if (i >= n4) return;
  float4 f = reinterpret_cast<const float4*>(src)[i];
  unsigned long long pk = (unsigned long long)f2bf(f.x) |
                          ((unsigned long long)f2bf(f.y) << 16) |
                          ((unsigned long long)f2bf(f.z) << 32) |
                          ((unsigned long long)f2bf(f.w) << 48);
  reinterpret_cast<unsigned long long*>(dst)[i] = pk;
}

// ---------------------------------------------------------------- GEMM: Y = X @ W^T (+epilogue)
// mode 0: Q proj -> RoPE, *1/sqrt(HD), store bf16 (b,h,s,hd)
// mode 1: K proj -> RoPE, store bf16 (b,h,s,hd)
// mode 2: V proj -> store bf16 (b,h,s,hd)
// mode 3: O proj -> +bias, store f32 (m,n)
__global__ __launch_bounds__(128) void gemm_qkvo(
    const unsigned short* __restrict__ X, const unsigned short* __restrict__ W0,
    const unsigned short* __restrict__ W1, const unsigned short* __restrict__ W2,
    unsigned short* __restrict__ dQ, unsigned short* __restrict__ dK,
    unsigned short* __restrict__ dV, float* __restrict__ dO,
    const float* __restrict__ bias, int baseMode) {
  const int mode = baseMode + (int)blockIdx.z;
  const unsigned short* W = (mode == 1) ? W1 : (mode == 2) ? W2 : W0;
  const int lane = (int)threadIdx.x & 31;
  const int wave = (int)threadIdx.x >> 5;
  const int m0 = (int)blockIdx.x * 64 + wave * 16;  // 4 waves x 16 rows
  const int n0 = (int)blockIdx.y * 64;              // 64 cols (== one head for modes 0-2)

  v8f acc[4] = {};
  for (int k0 = 0; k0 < Dv; k0 += 32) {
    v16bf a = load_a_global(X, Dv, m0, k0, lane);
#pragma unroll
    for (int j = 0; j < 4; ++j) {
      v16bf b = load_b_rowk(W, Dv, n0 + j * 16, k0, lane);
      acc[j] = wmma_bf16(a, b, acc[j]);
    }
  }

  const int half = lane >> 4;
  const int col = lane & 15;
  const int mbase = m0 + half * 8;

  if (mode == 3) {
#pragma unroll
    for (int r = 0; r < 8; ++r) {
      const size_t row = (size_t)(mbase + r) * Dv + n0;
#pragma unroll
      for (int j = 0; j < 4; ++j)
        dO[row + j * 16 + col] = acc[j][r] + bias[n0 + j * 16 + col];
    }
  } else if (mode == 2) {
#pragma unroll
    for (int r = 0; r < 8; ++r) {
      const int m = mbase + r;
      const size_t off =
          (((size_t)(m >> 11) * Hv + blockIdx.y) * Sv + (m & (Sv - 1))) * HDv;
#pragma unroll
      for (int j = 0; j < 4; ++j) dV[off + j * 16 + col] = f2bf(acc[j][r]);
    }
  } else {
    unsigned short* dst = mode ? dK : dQ;
    const float scl = mode ? 1.0f : 0.125f;  // fold 1/sqrt(64) into Q
    const float L = 9.210340371976184f;      // ln(10000)
    const float i0 = __expf(-L * (float)col * (1.0f / 32.0f));
    const float i1 = __expf(-L * (float)(col + 16) * (1.0f / 32.0f));
#pragma unroll
    for (int r = 0; r < 8; ++r) {
      const int m = mbase + r;
      const int s = m & (Sv - 1);
      const size_t off = (((size_t)(m >> 11) * Hv + blockIdx.y) * Sv + s) * HDv;
      const float fs = (float)s;
      const float a0 = fs * i0, a1 = fs * i1;
      const float c0 = __cosf(a0), s0 = __sinf(a0);
      const float c1 = __cosf(a1), s1 = __sinf(a1);
#pragma unroll
      for (int j = 0; j < 4; ++j) {  // hd = j*16+col ; partner hd^32 lives in acc[j^2]
        const float cv = (j & 1) ? c1 : c0;
        const float sv = (j & 1) ? s1 : s0;
        const float x = acc[j][r];
        const float xr = (j < 2) ? -acc[j ^ 2][r] : acc[j ^ 2][r];
        dst[off + j * 16 + col] = f2bf((x * cv + xr * sv) * scl);
      }
    }
  }
}

// ---------------------------------------------------------------- flash attention (causal)
__global__ __launch_bounds__(128) void attn_flash(
    const unsigned short* __restrict__ Q, const unsigned short* __restrict__ K,
    const unsigned short* __restrict__ V, unsigned short* __restrict__ Ctx) {
  extern __shared__ __align__(32) unsigned short smem[];
  unsigned short* Kt = smem;                       // [key][hd]  32 x KSTR (TDM dest)
  unsigned short* Vt = smem + 32 * KSTR;           // [hd][key]  64 x VSTR (transposed)
  unsigned short* Pl = smem + 32 * KSTR + 64 * VSTR;  // per-wave P, 4 x 16 x PSTR
  const unsigned int kt_lds_off = __builtin_amdgcn_groupstaticsize();  // dyn-LDS base

  const int lane = (int)threadIdx.x & 31;
  const int wave = (int)threadIdx.x >> 5;
  const int bh = (int)blockIdx.y;            // b*H + h
  const int q0 = (int)blockIdx.x * 64;       // 64 query rows per block
  const int qw = q0 + wave * 16;             // this wave's 16 query rows
  const size_t base = (size_t)bh * Sv * HDv;
  const unsigned short* Qb = Q + base;
  const unsigned short* Kb = K + base;
  const unsigned short* Vb = V + base;

  const v16bf aq0 = load_a_global(Qb, HDv, qw, 0, lane);
  const v16bf aq1 = load_a_global(Qb, HDv, qw, 32, lane);

  v8f co[4] = {};
  float mrow[8], lrow[8];
#pragma unroll
  for (int r = 0; r < 8; ++r) { mrow[r] = -3.0e38f; lrow[r] = 0.0f; }

  const int half = lane >> 4;
  const int col = lane & 15;
  const int qrow = qw + half * 8;

  const int t = (int)threadIdx.x;
  const int srow = t >> 2;          // 0..31 : key row to stage
  const int scg = (t & 3) << 4;     // 0,16,32,48 : hd group

  const int nkb = (q0 >> 5) + 2;    // key blocks of 32 covering keys <= q0+63
  for (int kb = 0; kb < nkb; ++kb) {
    const int kbase = kb * 32;
    __syncthreads();  // previous tile fully consumed
    // K tile via Tensor Data Mover (one wave issues the DMA; pad gives KSTR rows)
    if (wave == 0) tdm_load_k_tile(Kb + (size_t)kbase * HDv, kt_lds_off);
    {
      // stage V tile transposed : 16 scalar ds stores / thread
      const unsigned short* vsrc = Vb + (size_t)(kbase + srow) * HDv + scg;
#pragma unroll
      for (int i = 0; i < 16; ++i) Vt[(scg + i) * VSTR + srow] = vsrc[i];
      if (kb + 1 < nkb) {
        __builtin_prefetch(Kb + (size_t)(kbase + 32 + srow) * HDv + scg, 0, 1);
        __builtin_prefetch(Vb + (size_t)(kbase + 32 + srow) * HDv + scg, 0, 1);
      }
    }
    if (wave == 0) __builtin_amdgcn_s_wait_tensorcnt(0);
    __syncthreads();  // publishes TDM K tile + V^T stores to all waves

    // scores: two 16(q) x 16(key) tiles, contraction over hd=64 (2 WMMAs each)
    v8f s0 = {}, s1 = {};
    {
      const v16bf bk00 = *reinterpret_cast<const v16bf*>(&Kt[col * KSTR + half * 16]);
      const v16bf bk01 = *reinterpret_cast<const v16bf*>(&Kt[col * KSTR + 32 + half * 16]);
      const v16bf bk10 = *reinterpret_cast<const v16bf*>(&Kt[(16 + col) * KSTR + half * 16]);
      const v16bf bk11 = *reinterpret_cast<const v16bf*>(&Kt[(16 + col) * KSTR + 32 + half * 16]);
      s0 = wmma_bf16(aq0, bk00, s0);
      s0 = wmma_bf16(aq1, bk01, s0);
      s1 = wmma_bf16(aq0, bk10, s1);
      s1 = wmma_bf16(aq1, bk11, s1);
    }

    // causal mask + online softmax (rows live in 16-lane halves)
    const int key0 = kbase + col;
    const int key1 = key0 + 16;
#pragma unroll
    for (int r = 0; r < 8; ++r) {
      const int qg = qrow + r;
      float x0 = (key0 <= qg) ? s0[r] : -3.0e38f;
      float x1 = (key1 <= qg) ? s1[r] : -3.0e38f;
      float bm = fmaxf(x0, x1);
      bm = fmaxf(bm, __shfl_xor(bm, 1, 32));
      bm = fmaxf(bm, __shfl_xor(bm, 2, 32));
      bm = fmaxf(bm, __shfl_xor(bm, 4, 32));
      bm = fmaxf(bm, __shfl_xor(bm, 8, 32));
      const float mn = fmaxf(mrow[r], bm);
      const float p0 = __expf(x0 - mn);
      const float p1 = __expf(x1 - mn);
      float sm = p0 + p1;
      sm += __shfl_xor(sm, 1, 32);
      sm += __shfl_xor(sm, 2, 32);
      sm += __shfl_xor(sm, 4, 32);
      sm += __shfl_xor(sm, 8, 32);
      const float alpha = __expf(mrow[r] - mn);
      lrow[r] = lrow[r] * alpha + sm;
      mrow[r] = mn;
#pragma unroll
      for (int j = 0; j < 4; ++j) co[j][r] *= alpha;
      const int prow = half * 8 + r;  // C-layout -> A-layout via LDS
      Pl[wave * 16 * PSTR + prow * PSTR + col] = f2bf(p0);
      Pl[wave * 16 * PSTR + prow * PSTR + 16 + col] = f2bf(p1);
    }

    // ctx += P(16x32) @ V(32x64) : 4 WMMAs
    {
      const unsigned short* pp =
          &Pl[wave * 16 * PSTR + (lane & 15) * PSTR + (half << 4)];
      v8bf plo = *reinterpret_cast<const v8bf*>(pp);
      v8bf phi = *reinterpret_cast<const v8bf*>(pp + 8);
      const v16bf ap = combine16(plo, phi);
#pragma unroll
      for (int j = 0; j < 4; ++j) {
        const v16bf bv =
            *reinterpret_cast<const v16bf*>(&Vt[(j * 16 + col) * VSTR + half * 16]);
        co[j] = wmma_bf16(ap, bv, co[j]);
      }
    }
  }

  // normalize, store ctx as (b, s, h, hd) == (b, s, D) bf16
  const int b = bh >> 4;
  const int h = bh & 15;
#pragma unroll
  for (int r = 0; r < 8; ++r) {
    const int qg = qrow + r;
    const float inv = 1.0f / lrow[r];
    const size_t off = ((size_t)b * Sv + qg) * Dv + h * HDv;
#pragma unroll
    for (int j = 0; j < 4; ++j) Ctx[off + j * 16 + col] = f2bf(co[j][r] * inv);
  }
}

// ---------------------------------------------------------------- host side
extern "C" void kernel_launch(void* const* d_in, const int* in_sizes, int n_in,
                              void* d_out, int out_size, void* d_ws, size_t ws_size,
                              hipStream_t stream) {
  (void)in_sizes; (void)n_in; (void)out_size; (void)ws_size;
  const float* embeds = (const float*)d_in[0];
  const float* Wq = (const float*)d_in[1];
  const float* Wk = (const float*)d_in[2];
  const float* Wv = (const float*)d_in[3];
  const float* Wo = (const float*)d_in[4];
  const float* bo = (const float*)d_in[5];
  float* out = (float*)d_out;

  const size_t nX = (size_t)Bv * Sv * Dv;  // 4,194,304
  const size_t nW = (size_t)Dv * Dv;       // 1,048,576
  unsigned short* Xbf = (unsigned short*)d_ws;
  unsigned short* Wqb = Xbf + nX;
  unsigned short* Wkb = Wqb + nW;
  unsigned short* Wvb = Wkb + nW;
  unsigned short* Wob = Wvb + nW;
  unsigned short* Qr = Wob + nW;   // (b,h,s,hd) bf16, RoPE'd + scaled
  unsigned short* Kr = Qr + nX;    // (b,h,s,hd) bf16, RoPE'd
  unsigned short* Vr = Kr + nX;    // (b,h,s,hd) bf16
  unsigned short* Ctx = Vr + nX;   // (b,s,D)   bf16

  cvt_bf16<<<(int)(nX / 4 / 256), 256, 0, stream>>>(embeds, Xbf, (int)(nX / 4));
  cvt_bf16<<<(int)(nW / 4 / 256), 256, 0, stream>>>(Wq, Wqb, (int)(nW / 4));
  cvt_bf16<<<(int)(nW / 4 / 256), 256, 0, stream>>>(Wk, Wkb, (int)(nW / 4));
  cvt_bf16<<<(int)(nW / 4 / 256), 256, 0, stream>>>(Wv, Wvb, (int)(nW / 4));
  cvt_bf16<<<(int)(nW / 4 / 256), 256, 0, stream>>>(Wo, Wob, (int)(nW / 4));

  dim3 g1(Bv * Sv / 64, Dv / 64, 3);   // fused QKV projections + RoPE
  gemm_qkvo<<<g1, 128, 0, stream>>>(Xbf, Wqb, Wkb, Wvb, Qr, Kr, Vr, nullptr,
                                    nullptr, 0);

  dim3 g2(Sv / 64, Bv * Hv, 1);        // flash attention
  const size_t smemB = (size_t)(32 * KSTR + 64 * VSTR + 4 * 16 * PSTR) *
                       sizeof(unsigned short);  // 16384 B
  attn_flash<<<g2, 128, smemB, stream>>>(Qr, Kr, Vr, Ctx);

  dim3 g3(Bv * Sv / 64, Dv / 64, 1);   // output projection + bias
  gemm_qkvo<<<g3, 128, 0, stream>>>(Ctx, Wob, Wob, Wob, nullptr, nullptr,
                                    nullptr, out, bo, 3);
}
